// FocalLoss_137438953739
// MI455X (gfx1250) — matrix operand order repaired and checked by reading, not compiled, not run on hardware
//
// CDNA5 / gfx1250 fused focal-loss (linear + log-softmax + focal CE), compute-bound GEMM
// v2: software-pipelined (register double-buffered) WMMA K-loop + 32M x 64N wave tile
// (block tile 128M x 128N) to double arithmetic intensity and halve L2 W-traffic.
//
// Strategy: fp32->f16 convert once (f16 W = 103MB fits in 192MB L2), then
// v_wmma_f32_16x16x32_f16 GEMM with fused online-softmax strip reduction (never
// materializes the 824MB logits tensor), exact fp32 label-logit dots, per-row
// merge + focal weighting, loss = sum(w*ce)/sum(w).
//
// Workspace layout (total ~131 MB, assumed <= ws_size):
//   X16      : 4096*1024 f16                   =   8,388,608 B
//   W16      : 50304*1024 f16 (V padded, 0s)   = 103,022,592 B
//   partials : 4096 rows * 786 strips * float2 =  25,755,648 B
//   zlab     : 4096 f32                        =      16,384 B
//   acc      : 2 f32 (sum w*ce, sum w)

#include <hip/hip_runtime.h>
#include <math.h>

#define BB 2
#define SS 2048
#define DD 1024
#define VV 50257
#define RR (BB * SS)            // 4096 logits rows (last row per batch unused)
#define NTILE 393               // ceil(V / 128)
#define VPAD (NTILE * 128)      // 50304 (padded W rows, zero-filled)
#define NSTRIP (NTILE * 2)      // 786 per-row 64-column strips
#define IGNORE_IDX (-100)

typedef __attribute__((ext_vector_type(4)))  _Float16 v4h;
typedef __attribute__((ext_vector_type(8)))  _Float16 v8h;
typedef __attribute__((ext_vector_type(16))) _Float16 v16h;
typedef __attribute__((ext_vector_type(8)))  float    v8f;

// ---------------- fp32 -> f16 conversion passes ----------------

__global__ __launch_bounds__(256) void conv_x_f16(const float* __restrict__ X,
                                                  _Float16* __restrict__ X16) {
  size_t e = ((size_t)blockIdx.x * 256 + threadIdx.x) * 4;   // R*D = 4,194,304 total
  float4 x = *(const float4*)(X + e);
  v4h h; h[0] = (_Float16)x.x; h[1] = (_Float16)x.y; h[2] = (_Float16)x.z; h[3] = (_Float16)x.w;
  *(v4h*)(X16 + e) = h;
}

__global__ __launch_bounds__(256) void conv_w_f16(const float* __restrict__ W,
                                                  _Float16* __restrict__ W16) {
  size_t e = ((size_t)blockIdx.x * 256 + threadIdx.x) * 4;   // VPAD*D total
  size_t v = e >> 10;                                        // / DD
  v4h h;
  if (v < (size_t)VV) {
    float4 x = *(const float4*)(W + e);
    h[0] = (_Float16)x.x; h[1] = (_Float16)x.y; h[2] = (_Float16)x.z; h[3] = (_Float16)x.w;
  } else {
    h[0] = (_Float16)0.f; h[1] = (_Float16)0.f; h[2] = (_Float16)0.f; h[3] = (_Float16)0.f;
  }
  *(v4h*)(W16 + e) = h;
}

// ---------------- WMMA GEMM + fused per-strip online softmax ----------------
// Block = 256 threads = 8 waves in 4(M) x 2(N) -> 128M x 128N logits tile.
// Wave tile: 32M x 64N = 2 A-frags x 4 B-frags -> 8 v8f accumulators.
// K-loop over D=1024 in steps of 32, register double-buffered (prefetch next
// step's fragments before consuming current) so load latency overlaps WMMA.
// Epilogue: per (row, 64-col strip) write (max, sum_exp) partial.

// Build a 16x32 f16 A fragment (ISA layout): per-lane chunks at +0 and +16.
#define LOAD_A(dst, base, kk)                                           \
  do {                                                                  \
    v8h _a0 = *(const v8h*)((base) + (kk));                             \
    v8h _a1 = *(const v8h*)((base) + (kk) + 16);                        \
    _Pragma("unroll") for (int _i = 0; _i < 8; ++_i) {                  \
      (dst)[_i] = _a0[_i]; (dst)[_i + 8] = _a1[_i];                     \
    }                                                                   \
  } while (0)

// Build a 32x16 f16 B fragment (ISA layout): per-lane chunks at +0 and +16.
#define LOAD_B(dst, base, kk)                                           \
  do {                                                                  \
    v8h _b0 = *(const v8h*)((base) + (kk));                             \
    v8h _b1 = *(const v8h*)((base) + (kk) + 16);                        \
    _Pragma("unroll") for (int _i = 0; _i < 8; ++_i) {                  \
      (dst)[_i] = _b0[_i]; (dst)[_i + 8] = _b1[_i];                     \
    }                                                                   \
  } while (0)

__global__ __launch_bounds__(256) void gemm_softmax_partial(
    const _Float16* __restrict__ X16, const _Float16* __restrict__ W16,
    const float* __restrict__ bias, float2* __restrict__ partials) {
  const int lane = threadIdx.x & 31;
  const int wave = threadIdx.x >> 5;
  const int mw = wave & 3, nw = wave >> 2;
  const int n = lane & 15, h = lane >> 4;
  const int mBase = blockIdx.y * 128 + mw * 32;
  const int nBase = blockIdx.x * 128 + nw * 64;

  // A fragment rows: (mBase + mf*16 + n); K chunks at +h*8 / +h*8+16.
  const _Float16* aP[2];
#pragma unroll
  for (int mf = 0; mf < 2; ++mf)
    aP[mf] = X16 + (size_t)(mBase + mf * 16 + n) * DD + h * 8;
  // B fragment: W row = logits column; K chunks at +h*8 / +h*8+16.
  const _Float16* bP[4];
#pragma unroll
  for (int f = 0; f < 4; ++f)
    bP[f] = W16 + (size_t)(nBase + f * 16 + n) * DD + h * 8;

  v8f c[2][4] = {};
  v16h aBuf[2][2], bBuf[2][4];   // [pipeline stage][fragment]

  // prologue: stage 0 <- k = 0
#pragma unroll
  for (int mf = 0; mf < 2; ++mf) LOAD_A(aBuf[0][mf], aP[mf], 0);
#pragma unroll
  for (int f = 0; f < 4; ++f)    LOAD_B(bBuf[0][f], bP[f], 0);

  // steady state: 15 iterations of 2 K-steps, prefetching ahead
  for (int k = 0; k < DD - 64; k += 64) {
    // prefetch k+32 into stage 1
#pragma unroll
    for (int mf = 0; mf < 2; ++mf) LOAD_A(aBuf[1][mf], aP[mf], k + 32);
#pragma unroll
    for (int f = 0; f < 4; ++f)    LOAD_B(bBuf[1][f], bP[f], k + 32);
    // consume stage 0 (k)
#pragma unroll
    for (int mf = 0; mf < 2; ++mf)
#pragma unroll
      for (int f = 0; f < 4; ++f)
        c[mf][f] = __builtin_amdgcn_wmma_f32_16x16x32_f16(
            false, aBuf[0][mf], false, bBuf[0][f], (short)0, c[mf][f], false, false);
    // prefetch k+64 into stage 0
#pragma unroll
    for (int mf = 0; mf < 2; ++mf) LOAD_A(aBuf[0][mf], aP[mf], k + 64);
#pragma unroll
    for (int f = 0; f < 4; ++f)    LOAD_B(bBuf[0][f], bP[f], k + 64);
    // consume stage 1 (k+32)
#pragma unroll
    for (int mf = 0; mf < 2; ++mf)
#pragma unroll
      for (int f = 0; f < 4; ++f)
        c[mf][f] = __builtin_amdgcn_wmma_f32_16x16x32_f16(
            false, aBuf[1][mf], false, bBuf[1][f], (short)0, c[mf][f], false, false);
  }
  // epilogue of the pipeline: k = DD-64 and DD-32, no further prefetch
  {
#pragma unroll
    for (int mf = 0; mf < 2; ++mf) LOAD_A(aBuf[1][mf], aP[mf], DD - 32);
#pragma unroll
    for (int f = 0; f < 4; ++f)    LOAD_B(bBuf[1][f], bP[f], DD - 32);
#pragma unroll
    for (int mf = 0; mf < 2; ++mf)
#pragma unroll
      for (int f = 0; f < 4; ++f)
        c[mf][f] = __builtin_amdgcn_wmma_f32_16x16x32_f16(
            false, aBuf[0][mf], false, bBuf[0][f], (short)0, c[mf][f], false, false);
#pragma unroll
    for (int mf = 0; mf < 2; ++mf)
#pragma unroll
      for (int f = 0; f < 4; ++f)
        c[mf][f] = __builtin_amdgcn_wmma_f32_16x16x32_f16(
            false, aBuf[1][mf], false, bBuf[1][f], (short)0, c[mf][f], false, false);
  }

  // ---- Epilogue: per lane, 4 columns (nBase + f*16 + n); mask cols >= V ----
  float bv[4]; bool ok[4];
#pragma unroll
  for (int f = 0; f < 4; ++f) {
    int col = nBase + f * 16 + n;
    ok[f] = (col < VV);
    bv[f] = ok[f] ? bias[col] : 0.f;
  }

  const float NEG_INF = -__builtin_inff();
  const int strip = blockIdx.x * 2 + nw;
#pragma unroll
  for (int mf = 0; mf < 2; ++mf) {
#pragma unroll
    for (int r = 0; r < 8; ++r) {   // row = mBase + mf*16 + r + 8*h (C/D layout)
      float vals[4];
      float m = NEG_INF;
#pragma unroll
      for (int f = 0; f < 4; ++f) {
        vals[f] = ok[f] ? (c[mf][f][r] + bv[f]) : NEG_INF;
        m = fmaxf(m, vals[f]);
      }
      // reduce max over the 16 lanes sharing this row (xor < 16 stays in-group)
      for (int off = 1; off < 16; off <<= 1) m = fmaxf(m, __shfl_xor(m, off, 32));
      float s = 0.f;
#pragma unroll
      for (int f = 0; f < 4; ++f)
        if (ok[f]) s += __expf(vals[f] - m);
      for (int off = 1; off < 16; off <<= 1) s += __shfl_xor(s, off, 32);

      if (n == 0) {
        int rowG = mBase + mf * 16 + r + 8 * h;
        partials[(size_t)rowG * NSTRIP + strip] = make_float2(m, s);
      }
    }
  }
}

// ---------------- exact fp32 label logits (one wave per row) ----------------

__global__ __launch_bounds__(256) void label_logit(
    const float* __restrict__ X, const float* __restrict__ W,
    const float* __restrict__ bias, const int* __restrict__ labels,
    float* __restrict__ zlab) {
  int wave = threadIdx.x >> 5, lane = threadIdx.x & 31;
  int r = blockIdx.x * 8 + wave;
  if (r >= RR) return;
  int s = r & (SS - 1), b_ = r >> 11;   // SS = 2048 = 2^11
  float z = 0.f;
  int li = IGNORE_IDX;
  if (s < SS - 1) li = labels[b_ * SS + s + 1];
  if (li >= 0 && li < VV) {
    const float4* x = (const float4*)(X + (size_t)r * DD);
    const float4* w = (const float4*)(W + (size_t)li * DD);
    float acc = 0.f;
    for (int i = lane; i < DD / 4; i += 32) {
      float4 a = x[i], c = w[i];
      acc += a.x * c.x + a.y * c.y + a.z * c.z + a.w * c.w;
    }
    for (int off = 16; off; off >>= 1) acc += __shfl_xor(acc, off, 32);
    z = acc + bias[li];
  }
  if (lane == 0) zlab[r] = z;
}

// ---------------- per-row merge + focal accumulation ----------------

__global__ __launch_bounds__(256) void row_reduce(
    const float2* __restrict__ partials, const float* __restrict__ zlab,
    const int* __restrict__ labels, float* __restrict__ acc) {
  int wave = threadIdx.x >> 5, lane = threadIdx.x & 31;
  int r = blockIdx.x * 8 + wave;
  if (r >= RR) return;
  int s = r & (SS - 1), b_ = r >> 11;
  if (s >= SS - 1) return;                        // shifted-off row
  int li = labels[b_ * SS + s + 1];
  if (li < 0 || li >= VV) return;                 // ignore_index -> w=0, ce=0

  const float NEG_INF = -__builtin_inff();
  float m = NEG_INF, z = 0.f;
  const float2* p = partials + (size_t)r * NSTRIP;
  for (int t = lane; t < NSTRIP; t += 32) {
    float2 q = p[t];
    if (q.y > 0.f) {
      float nm = fmaxf(m, q.x);
      z = z * __expf(m - nm) + q.y * __expf(q.x - nm);
      m = nm;
    }
  }
  for (int off = 16; off; off >>= 1) {
    float m2 = __shfl_xor(m, off, 32);
    float z2 = __shfl_xor(z, off, 32);
    if (z2 > 0.f) {
      float nm = fmaxf(m, m2);
      z = z * __expf(m - nm) + z2 * __expf(m2 - nm);
      m = nm;
    }
  }
  if (lane == 0) {
    float ce = m + __logf(z) - zlab[r];           // -log p(label)
    float w = 1.f - __expf(-ce);                  // focal weight, gamma = 1
    atomicAdd(&acc[0], w * ce);
    atomicAdd(&acc[1], w);
  }
}

__global__ void init_acc(float* acc) { acc[0] = 0.f; acc[1] = 0.f; }

__global__ void finalize(const float* __restrict__ acc, float* __restrict__ out) {
  float d = acc[1];
  out[0] = (d != 0.f) ? (acc[0] / d) : 0.f;       // sum(w*ce)/sum(w) == alpha*mean(w*ce)
}

// ---------------- host side ----------------

extern "C" void kernel_launch(void* const* d_in, const int* in_sizes, int n_in,
                              void* d_out, int out_size, void* d_ws, size_t ws_size,
                              hipStream_t stream) {
  const float* X    = (const float*)d_in[0];   // [B,S,D]
  const float* W    = (const float*)d_in[1];   // [V,D]
  const float* bias = (const float*)d_in[2];   // [V]
  const int*   lab  = (const int*)d_in[3];     // [B,S]
  float* out = (float*)d_out;

  char* wsb = (char*)d_ws;
  size_t offX16  = 0;
  size_t offW16  = offX16 + (size_t)RR * DD * sizeof(_Float16);     //   8,388,608
  size_t offPart = offW16 + (size_t)VPAD * DD * sizeof(_Float16);   // 111,411,200
  size_t offZlab = offPart + (size_t)RR * NSTRIP * sizeof(float2);  // 137,166,848
  size_t offAcc  = offZlab + (size_t)RR * sizeof(float);            // 137,183,232
  (void)ws_size; (void)in_sizes; (void)n_in; (void)out_size;

  _Float16* X16 = (_Float16*)(wsb + offX16);
  _Float16* W16 = (_Float16*)(wsb + offW16);
  float2* partials = (float2*)(wsb + offPart);
  float* zlab = (float*)(wsb + offZlab);
  float* acc  = (float*)(wsb + offAcc);

  init_acc<<<1, 1, 0, stream>>>(acc);

  conv_x_f16<<<(RR * DD) / (256 * 4), 256, 0, stream>>>(X, X16);
  conv_w_f16<<<((size_t)VPAD * DD) / (256 * 4), 256, 0, stream>>>(W, W16);

  dim3 grid(NTILE, RR / 128);   // 393 x 32 tiles of 128(V) x 128(rows)
  gemm_softmax_partial<<<grid, 256, 0, stream>>>(X16, W16, bias, partials);

  label_logit<<<RR / 8, 256, 0, stream>>>(X, W, bias, lab, zlab);
  row_reduce<<<RR / 8, 256, 0, stream>>>(partials, zlab, lab, acc);
  finalize<<<1, 1, 0, stream>>>(acc, out);
}
